// BCNN_73254962200863
// MI455X (gfx1250) — compile-verified, matrix operand order checked by reading, not run on hardware
//
#include <hip/hip_runtime.h>
#include <hip/hip_bf16.h>
#include <math.h>

// ---------------------------------------------------------------------------
// BCNN sentence scorer, algebraically collapsed.
//
// All conv outputs are consumed only through global means (no nonlinearity),
// so:   mean(conv1(x)) = b1 + (W1s @ S)/515,  W1s[o,i] = sum_k w1[o,i,k],
//       S[n,i] = sum_l x[n,i,l]
//       T[n]   = sum_t windowpool(conv1(x))   (needed for mean(conv2(...)))
//              = 512*b1 + W1s@S[n] - 0.25 * sum_m A_m @ x[n,:,l_m]
//         (A_m = boundary tap combos of w1, l_m in {0,1,2,L-3,L-2,L-1})
//       mean(conv2(pool)) = b2 + (W2s @ T)/515
// Question side identical with L=32, denominators 35.
// => one HBM streaming pass (240 MB, ~11us at 23.3 TB/s) + tiny WMMA GEMMs.
// ---------------------------------------------------------------------------

typedef __attribute__((ext_vector_type(16))) _Float16 v16h;
typedef __attribute__((ext_vector_type(8)))  float    v8f;

#define NS   4096
#define LS   512
#define LQ   32
#define DD   30
#define DP   32

// workspace layout (floats)
#define WS_W1S   0            // [32*32]
#define WS_W2S   1024         // [32*32]
#define WS_AM    2048         // [6][32*32]
#define WS_QRAW  8192         // [32]
#define WS_Q1G   8224         // [32]
#define WS_Q2G   8256         // [32]
#define WS_S     8320         // [4096][32]
#define WS_PART  (8320 + 4096*32)   // [64]

// A/B fragment element -> K index (ISA 16-bit 16x32 A layout, wave32)
#define KMAP(e, lane) (((e) < 8) ? ((((lane) >> 4) << 3) + (e)) \
                                 : (16 + (((lane) >> 4) << 3) + ((e) - 8)))

static __device__ inline v8f wmma_f16(v16h a, v16h b, v8f c) {
  return __builtin_amdgcn_wmma_f32_16x16x32_f16(
      false, a, false, b, (short)0, c, false, false);
}

// B[k][col] = M[(tile*16+col)*32 + k] ; M is a zero-padded 32x32 f32 matrix
static __device__ inline v16h load_bfrag(const float* M, int tile, int lane) {
  const int col = lane & 15;
  const float* base = M + (tile * 16 + col) * 32;
  v16h b;
#pragma unroll
  for (int e = 0; e < 16; ++e) b[e] = (_Float16)base[KMAP(e, lane)];
  return b;
}

// ---------------------------------------------------------------------------
// Kernel 0: fold conv weights, compute question-side vectors (1 block, 256 thr)
// ---------------------------------------------------------------------------
__global__ void bcnn_prep(const float* __restrict__ q,
                          const float* __restrict__ w1, const float* __restrict__ b1,
                          const float* __restrict__ w2, const float* __restrict__ b2,
                          float* __restrict__ ws) {
  float* W1s  = ws + WS_W1S;
  float* W2s  = ws + WS_W2S;
  float* Am   = ws + WS_AM;
  float* qraw = ws + WS_QRAW;
  float* q1g  = ws + WS_Q1G;
  float* q2g  = ws + WS_Q2G;
  const int tid = threadIdx.x;

  for (int idx = tid; idx < 1024; idx += 256) {
    const int o = idx >> 5, i = idx & 31;
    const bool v = (o < DD) && (i < DD);
    float a0 = 0.f, a1 = 0.f, a2 = 0.f, a3 = 0.f;
    float c0 = 0.f, c1 = 0.f, c2 = 0.f, c3 = 0.f;
    if (v) {
      const float* p1 = w1 + (o * DD + i) * 4;
      const float* p2 = w2 + (o * DD + i) * 4;
      a0 = p1[0]; a1 = p1[1]; a2 = p1[2]; a3 = p1[3];
      c0 = p2[0]; c1 = p2[1]; c2 = p2[2]; c3 = p2[3];
    }
    W1s[idx] = a0 + a1 + a2 + a3;
    W2s[idx] = c0 + c1 + c2 + c3;
    Am[0 * 1024 + idx] = 3.f * a3 + 2.f * a2 + a1;  // x[0]
    Am[1 * 1024 + idx] = 2.f * a3 + a2;             // x[1]
    Am[2 * 1024 + idx] = a3;                        // x[2]
    Am[3 * 1024 + idx] = a0;                        // x[L-3]
    Am[4 * 1024 + idx] = 2.f * a0 + a1;             // x[L-2]
    Am[5 * 1024 + idx] = 3.f * a0 + 2.f * a1 + a2;  // x[L-1]
  }

  __shared__ float qs[32], tq[32];
  __syncthreads();

  if (tid < 32) {
    const int i = tid;
    float s = 0.f;
    if (i < DD)
      for (int l = 0; l < LQ; ++l) s += q[l * DD + i];
    qs[i] = s;
    qraw[i] = (i < DD) ? s * (1.f / 32.f) : 0.f;
  }
  __syncthreads();

  if (tid < 32) {
    const int o = tid;
    float Uq = 0.f;
    for (int i = 0; i < 32; ++i) Uq += W1s[o * 32 + i] * qs[i];
    float Eq = 0.f;
    const int lq[6] = {0, 1, 2, LQ - 3, LQ - 2, LQ - 1};
    for (int m = 0; m < 6; ++m)
      for (int j = 0; j < DD; ++j)
        Eq += Am[m * 1024 + o * 32 + j] * q[lq[m] * DD + j];
    const float b1o = (o < DD) ? b1[o] : 0.f;
    q1g[o] = (o < DD) ? b1o + Uq * (1.f / 35.f) : 0.f;
    tq[o]  = (o < DD) ? 32.f * b1o + Uq - 0.25f * Eq : 0.f;
  }
  __syncthreads();

  if (tid < 32) {
    const int o = tid;
    float r = 0.f;
    for (int i = 0; i < 32; ++i) r += W2s[o * 32 + i] * tq[i];
    q2g[o] = (o < DD) ? b2[o] + r * (1.f / 35.f) : 0.f;
  }
}

// ---------------------------------------------------------------------------
// Kernel 1: per-sentence channel sums S[n][i] — the only HBM-heavy pass.
// float4 loads: 4 rows = 120 floats == 0 (mod 30), so the float4 at offset
// g*120 + 4*lane always covers channels (4*lane+t) % 30 — fixed per thread.
// Fully coalesced 480B wave-loads; deterministic LDS unscramble at the end.
// ---------------------------------------------------------------------------
__global__ void bcnn_sums(const float* __restrict__ sents, float* __restrict__ Sbuf) {
  const int n    = blockIdx.x;
  const int lane = threadIdx.x & 31;
  const int wv   = threadIdx.x >> 5;
  const float4* sp = (const float4*)(sents + (size_t)n * (LS * DD));

  float a0 = 0.f, a1 = 0.f, a2 = 0.f, a3 = 0.f;
  if (lane < DD) {
    const int g0 = wv * 16, g1 = g0 + 16;   // 128 groups of 4 rows, 16 per wave
    for (int g = g0; g < g1; ++g) {
      const float4 v = sp[g * 30 + lane];
      a0 += v.x; a1 += v.y; a2 += v.z; a3 += v.w;
    }
  }
  __shared__ float red[8][30][4];
  if (lane < DD) {
    red[wv][lane][0] = a0; red[wv][lane][1] = a1;
    red[wv][lane][2] = a2; red[wv][lane][3] = a3;
  }
  __syncthreads();
  if (wv == 0) {
    float t = 0.f;
    if (lane < DD) {
#pragma unroll
      for (int p = 0; p < 4; ++p) {          // slots {c, c+30, c+60, c+90}
        const int v = lane + 30 * p;
        const int j = v >> 2, tt = v & 3;
#pragma unroll
        for (int w = 0; w < 8; ++w) t += red[w][j][tt];
      }
    }
    Sbuf[n * DP + lane] = (lane < DD) ? t : 0.f;  // zero padding in cols 30,31
  }
}

// ---------------------------------------------------------------------------
// Kernel 2: WMMA feature GEMMs + similarities + MLP + emit + partial cost.
// 64 blocks x 128 threads; each wave owns 16 sentences (block owns 64).
// ---------------------------------------------------------------------------
__global__ void bcnn_feat(const float* __restrict__ sents,
                          const float* __restrict__ gaf,
                          const int*   __restrict__ labels,
                          const float* __restrict__ b1, const float* __restrict__ b2,
                          const float* __restrict__ lw, const float* __restrict__ lb,
                          float* __restrict__ ws, float* __restrict__ out) {
  const float* W1s  = ws + WS_W1S;
  const float* W2s  = ws + WS_W2S;
  const float* Am   = ws + WS_AM;
  const float* qraw = ws + WS_QRAW;
  const float* q1g  = ws + WS_Q1G;
  const float* q2g  = ws + WS_Q2G;
  const float* Sbuf = ws + WS_S;

  __shared__ _Float16 Tlds[4][16][32];
  __shared__ float s1gl[64][32];
  __shared__ float s2gl[64][32];
  __shared__ float red[128];

  const int tid  = threadIdx.x;
  const int wv   = tid >> 5;
  const int lane = tid & 31;
  const int row  = lane & 15;
  const int hi   = lane >> 4;
  const int col  = lane & 15;
  const int n0   = blockIdx.x * 64 + wv * 16;

  const v8f z = {0.f, 0.f, 0.f, 0.f, 0.f, 0.f, 0.f, 0.f};

  // --- GEMM1: U = S @ W1s^T  (16 sentences x 32 channels, K=32) ---
  v16h aS;
  {
    const float* base = Sbuf + (size_t)(n0 + row) * DP;
#pragma unroll
    for (int e = 0; e < 16; ++e) aS[e] = (_Float16)base[KMAP(e, lane)];
  }
  v8f u[2];  u[0] = z; u[1] = z;
#pragma unroll
  for (int t = 0; t < 2; ++t)
    u[t] = wmma_f16(aS, load_bfrag(W1s, t, lane), u[t]);

  // --- Edge-correction GEMM: E = sum_m xe_m @ A_m^T  (K=6x32) ---
  // Branchless masking: unconditional load at clamped index + value select.
  v8f ce[2]; ce[0] = z; ce[1] = z;
  const int EL[6] = {0, 1, 2, LS - 3, LS - 2, LS - 1};
#pragma unroll
  for (int m = 0; m < 6; ++m) {
    v16h aE;
    const float* base = sents + ((size_t)(n0 + row) * LS + EL[m]) * DD;
#pragma unroll
    for (int e = 0; e < 16; ++e) {
      const int k  = KMAP(e, lane);
      const float v = base[(k < DD) ? k : (DD - 1)];
      aE[e] = (_Float16)((k < DD) ? v : 0.f);
    }
#pragma unroll
    for (int t = 0; t < 2; ++t)
      ce[t] = wmma_f16(aE, load_bfrag(Am + m * 1024, t, lane), ce[t]);
  }

  // --- T = 512*b1 + U - 0.25*E ; s1g = b1 + U/515 (C-fragment layout) ---
#pragma unroll
  for (int t = 0; t < 2; ++t) {
    const int i = t * 16 + col;
    const float b1v = b1[(i < DD) ? i : (DD - 1)];
    const float b1i = (i < DD) ? b1v : 0.f;
#pragma unroll
    for (int r = 0; r < 8; ++r) {
      const int nl = r + 8 * hi;
      const float Uv = u[t][r], Ev = ce[t][r];
      Tlds[wv][nl][i]       = (_Float16)(512.f * b1i + Uv - 0.25f * Ev);
      s1gl[wv * 16 + nl][i] = b1i + Uv * (1.f / 515.f);
    }
  }
  __syncthreads();

  // --- GEMM2: R2 = T @ W2s^T ---
  v16h aT;
#pragma unroll
  for (int e = 0; e < 16; ++e) aT[e] = Tlds[wv][row][KMAP(e, lane)];
  v8f r2[2]; r2[0] = z; r2[1] = z;
#pragma unroll
  for (int t = 0; t < 2; ++t)
    r2[t] = wmma_f16(aT, load_bfrag(W2s, t, lane), r2[t]);

#pragma unroll
  for (int t = 0; t < 2; ++t) {
    const int i = t * 16 + col;
    const float b2v = b2[(i < DD) ? i : (DD - 1)];
    const float b2i = (i < DD) ? b2v : 0.f;
#pragma unroll
    for (int r = 0; r < 8; ++r) {
      const int nl = r + 8 * hi;
      s2gl[wv * 16 + nl][i] = b2i + r2[t][r] * (1.f / 515.f);
    }
  }
  __syncthreads();

  // --- Per-sentence cosines, MLP, log-softmax, emit, cost ---
  float closs = 0.f;
  if (tid < 64) {
    const int n = blockIdx.x * 64 + tid;
    float d1 = 0.f, nq1 = 0.f, ns1 = 0.f;
    float d2 = 0.f, nq2 = 0.f, ns2 = 0.f;
    float d3 = 0.f, nq3 = 0.f, ns3 = 0.f;
    for (int i = 0; i < DD; ++i) {
      const float sr = Sbuf[(size_t)n * DP + i] * (1.f / 512.f);
      const float qv = (n == 0) ? qraw[i] : q2g[i];
      d1 += qv * sr;  nq1 += qv * qv;  ns1 += sr * sr;
      const float s1 = s1gl[tid][i], qa = q1g[i];
      d2 += qa * s1;  nq2 += qa * qa;  ns2 += s1 * s1;
      const float s2 = s2gl[tid][i], qb = q2g[i];
      d3 += qb * s2;  nq3 += qb * qb;  ns3 += s2 * s2;
    }
    const float sim1 = d1 / (sqrtf(nq1) * sqrtf(ns1));
    const float sim2 = d2 / (sqrtf(nq2) * sqrtf(ns2));
    const float sim3 = d3 / (sqrtf(nq3) * sqrtf(ns3));

    float l0 = lb[0] + lw[0]  * sim1 + lw[1]  * sim2 + lw[2]  * sim3;
    float l1 = lb[1] + lw[13] * sim1 + lw[14] * sim2 + lw[15] * sim3;
    for (int f = 0; f < 10; ++f) {
      const float g = gaf[n * 10 + f];
      l0 += lw[3 + f] * g;
      l1 += lw[16 + f] * g;
    }
    const float mx  = fmaxf(l0, l1);
    const float lse = mx + logf(expf(l0 - mx) + expf(l1 - mx));
    const float lp0 = l0 - lse, lp1 = l1 - lse;
    out[1 + n] = expf(lp1);                 // softmax(log_softmax)[.,1] == exp(logp1)
    closs = (labels[n] != 0) ? -lp1 : -lp0;
  }
  red[tid] = closs;
  __syncthreads();
#pragma unroll
  for (int s = 64; s >= 1; s >>= 1) {
    if (tid < s) red[tid] += red[tid + s];
    __syncthreads();
  }
  if (tid == 0) ws[WS_PART + blockIdx.x] = red[0];
}

// ---------------------------------------------------------------------------
// Kernel 3: deterministic final cost reduction.
// ---------------------------------------------------------------------------
__global__ void bcnn_cost(const float* __restrict__ ws, float* __restrict__ out) {
  if (threadIdx.x == 0) {
    float s = 0.f;
    for (int b = 0; b < 64; ++b) s += ws[WS_PART + b];
    out[0] = s * (1.f / (float)NS);
  }
}

extern "C" void kernel_launch(void* const* d_in, const int* in_sizes, int n_in,
                              void* d_out, int out_size, void* d_ws, size_t ws_size,
                              hipStream_t stream) {
  (void)in_sizes; (void)n_in; (void)out_size; (void)ws_size;
  const float* sents  = (const float*)d_in[0];
  const float* q      = (const float*)d_in[1];
  const float* gaf    = (const float*)d_in[2];
  const int*   labels = (const int*)d_in[3];
  const float* w1     = (const float*)d_in[4];
  const float* b1     = (const float*)d_in[5];
  const float* w2     = (const float*)d_in[6];
  const float* b2     = (const float*)d_in[7];
  const float* lw     = (const float*)d_in[8];
  const float* lb     = (const float*)d_in[9];
  float* ws  = (float*)d_ws;
  float* out = (float*)d_out;

  bcnn_prep<<<1, 256, 0, stream>>>(q, w1, b1, w2, b2, ws);
  bcnn_sums<<<NS, 256, 0, stream>>>(sents, ws + WS_S);
  bcnn_feat<<<64, 128, 0, stream>>>(sents, gaf, labels, b1, b2, lw, lb, ws, out);
  bcnn_cost<<<1, 32, 0, stream>>>(ws, out);
}
